// NystromAttention_7988639170986
// MI455X (gfx1250) — matrix operand order repaired and checked by reading, not compiled, not run on hardware
//
#include <hip/hip_runtime.h>
#include <hip/hip_bf16.h>

// ---------------- problem constants ----------------
#define B_     4
#define N_     4096
#define DIM_   512
#define H_     8
#define DH_    64
#define M_     256
#define SEG_   16          // N_/M_
#define BH_    32          // B_*H_
#define INNER_ 512         // H_*DH_
#define KS_    33
#define PAD_   16
#define ITERS_ 6
#define SCALE_ 0.125f      // DH^-0.5

// LDS B-panel row stride (64 data DWORDs + 2 pad DWORDs from TDM pad feature)
#define BSTRIDE 66
#define KSTAGE  128        // K rows staged per TDM transfer

// ---------------- WMMA types (wave32, gfx1250) ----------------
typedef __bf16 bf16_t;
typedef bf16_t bf16x16 __attribute__((ext_vector_type(16)));
typedef float  f32x8   __attribute__((ext_vector_type(8)));
typedef float  f32x2   __attribute__((ext_vector_type(2)));
typedef unsigned int u32x4 __attribute__((ext_vector_type(4)));
typedef int    i32x8  __attribute__((ext_vector_type(8)));
typedef int    i32x4  __attribute__((ext_vector_type(4)));

__device__ __forceinline__ int lane_id() { return (int)(threadIdx.x & 31u); }

// D = A(16x32 bf16) * B(32x16 bf16) + C, f32 accum
__device__ __forceinline__ f32x8 wmma_bf16(bf16x16 a, bf16x16 b, f32x8 c) {
  return __builtin_amdgcn_wmma_f32_16x16x32_bf16(false, a, false, b, (short)0, c,
                                                 false, false);
}
// D = A(16x4 f32) * B(4x16 f32) + C  — full fp32 matrix op
__device__ __forceinline__ f32x8 wmma_f32(f32x2 a, f32x2 b, f32x8 c) {
  return __builtin_amdgcn_wmma_f32_16x16x4_f32(false, a, false, b, (short)0, c,
                                               false, false);
}

#if __has_builtin(__builtin_amdgcn_tensor_load_to_lds)
#define HAVE_TDM 1
#endif

// --------- tile loaders (layouts per CDNA5 ISA 7.12.2, wave32) ---------
__device__ __forceinline__ bf16x16 ld_a16(const bf16_t* a, int lda) {
  const int l = lane_id(), m = l & 15, h8 = (l >> 4) * 8;
  bf16x16 r;
#pragma unroll
  for (int j = 0; j < 8; ++j) {
    const int k = ((j & 3) * 2) + ((j >> 2) * 16) + h8;
    r[2 * j]     = a[(long)m * lda + k];
    r[2 * j + 1] = a[(long)m * lda + k + 1];
  }
  return r;
}
__device__ __forceinline__ bf16x16 ld_a16_f(const float* a, int lda) {
  const int l = lane_id(), m = l & 15, h8 = (l >> 4) * 8;
  bf16x16 r;
#pragma unroll
  for (int j = 0; j < 8; ++j) {
    const int k = ((j & 3) * 2) + ((j >> 2) * 16) + h8;
    r[2 * j]     = (bf16_t)a[(long)m * lda + k];
    r[2 * j + 1] = (bf16_t)a[(long)m * lda + k + 1];
  }
  return r;
}
__device__ __forceinline__ bf16x16 ld_b16(const bf16_t* b, int ldb) {
  const int l = lane_id(), n = l & 15, h8 = (l >> 4) * 8;
  bf16x16 r;
#pragma unroll
  for (int j = 0; j < 8; ++j) {
    const int k = ((j & 3) * 2) + ((j >> 2) * 16) + h8;
    r[2 * j]     = b[(long)k * ldb + n];
    r[2 * j + 1] = b[(long)(k + 1) * ldb + n];
  }
  return r;
}
__device__ __forceinline__ bf16x16 ld_b16_f(const float* b, int ldb) {
  const int l = lane_id(), n = l & 15, h8 = (l >> 4) * 8;
  bf16x16 r;
#pragma unroll
  for (int j = 0; j < 8; ++j) {
    const int k = ((j & 3) * 2) + ((j >> 2) * 16) + h8;
    r[2 * j]     = (bf16_t)b[(long)k * ldb + n];
    r[2 * j + 1] = (bf16_t)b[(long)(k + 1) * ldb + n];
  }
  return r;
}
// B-transposed: logical B[k][n] = src[n][k], src row-major [Ncols x K]
__device__ __forceinline__ bf16x16 ld_bt16(const bf16_t* s, int lds_) {
  const int l = lane_id(), n = l & 15, h8 = (l >> 4) * 8;
  bf16x16 r;
#pragma unroll
  for (int j = 0; j < 8; ++j) {
    const int k = ((j & 3) * 2) + ((j >> 2) * 16) + h8;
    r[2 * j]     = s[(long)n * lds_ + k];
    r[2 * j + 1] = s[(long)n * lds_ + k + 1];
  }
  return r;
}
// fp32 WMMA A 16x4: VGPR j -> K = j + 2*half
__device__ __forceinline__ f32x2 ld_a4(const float* a, int lda) {
  const int l = lane_id(), m = l & 15, kh = (l >> 4) * 2;
  f32x2 r; r[0] = a[(long)m * lda + kh]; r[1] = a[(long)m * lda + kh + 1];
  return r;
}

// ================= kernels =================

// QKV projection: X[B*N,512] @ Wqkv[512,1536] + b; split/scale/scatter.
__global__ __launch_bounds__(256) void k_qkv(const float* __restrict__ x,
                                             const float* __restrict__ w,
                                             const float* __restrict__ bias,
                                             bf16_t* __restrict__ Qb,
                                             bf16_t* __restrict__ Kb,
                                             bf16_t* __restrict__ Vb,
                                             float* __restrict__ Vf) {
  const int wave = threadIdx.x >> 5;
  const int row0 = blockIdx.y * 32 + (wave >> 2) * 16;
  const int col0 = blockIdx.x * 64 + (wave & 3) * 16;
  f32x8 acc = {};
  for (int kk = 0; kk < DIM_; kk += 32) {
    __builtin_prefetch(x + (long)row0 * DIM_ + kk + 64, 0, 3);
    bf16x16 a = ld_a16_f(x + (long)row0 * DIM_ + kk, DIM_);
    bf16x16 b = ld_b16_f(w + (long)kk * (3 * INNER_) + col0, 3 * INNER_);
    acc = wmma_bf16(a, b, acc);
  }
  const int l = lane_id(), n = l & 15, mb = (l >> 4) * 8;
  const int c = col0 + n;
  const int sel = c >> 9, inner = c & 511;
  const int h = inner >> 6, d = inner & 63;
  const float bv = bias[c];
#pragma unroll
  for (int r = 0; r < 8; ++r) {
    const int gm = row0 + mb + r;
    const int bidx = gm >> 12, npos = gm & (N_ - 1);
    const long o = (((long)(bidx * H_ + h)) * N_ + npos) * DH_ + d;
    float v = acc[r] + bv;
    if (sel == 0)      Qb[o] = (bf16_t)(v * SCALE_);
    else if (sel == 1) Kb[o] = (bf16_t)v;
    else { Vf[o] = v; Vb[o] = (bf16_t)v; }
  }
}

// landmark pooling: mean of SEG_ consecutive rows
__global__ void k_landmark(const bf16_t* __restrict__ src, bf16_t* __restrict__ dst) {
  long i = (long)blockIdx.x * blockDim.x + threadIdx.x;
  if (i >= (long)BH_ * M_ * DH_) return;
  int d = (int)(i % DH_); long t = i / DH_;
  int m = (int)(t % M_); int bh = (int)(t / M_);
  const bf16_t* p = src + ((long)bh * N_ + (long)m * SEG_) * DH_ + d;
  float s = 0.f;
#pragma unroll
  for (int j = 0; j < SEG_; ++j) s += (float)p[(long)j * DH_];
  dst[i] = (bf16_t)(s * (1.0f / SEG_));
}

// batched C = A @ B^T ; bf16 in, bf16 (Ob) or f32 (Of) out
__global__ __launch_bounds__(256) void gemm_bf16_nt(
    const bf16_t* __restrict__ A, int lda, long sA,
    const bf16_t* __restrict__ Bm, int ldb, long sB,
    bf16_t* __restrict__ Ob, float* __restrict__ Of, int ldo, long sO, int Kd) {
  const int bh = blockIdx.z;
  const int wave = threadIdx.x >> 5;
  const int row0 = blockIdx.y * 32 + (wave >> 2) * 16;
  const int col0 = blockIdx.x * 64 + (wave & 3) * 16;
  const bf16_t* Ap = A + (long)bh * sA + (long)row0 * lda;
  const bf16_t* Bp = Bm + (long)bh * sB + (long)col0 * ldb;
  f32x8 acc = {};
  for (int kk = 0; kk < Kd; kk += 32) {
    __builtin_prefetch(Ap + kk + 32, 0, 3);
    acc = wmma_bf16(ld_a16(Ap + kk, lda), ld_bt16(Bp + kk, ldb), acc);
  }
  const int l = lane_id(), n = l & 15, mb = (l >> 4) * 8;
  const long base = (long)bh * sO + (long)row0 * ldo + col0;
  if (Ob) {
#pragma unroll
    for (int r = 0; r < 8; ++r) Ob[base + (long)(mb + r) * ldo + n] = (bf16_t)acc[r];
  } else {
#pragma unroll
    for (int r = 0; r < 8; ++r) Of[base + (long)(mb + r) * ldo + n] = acc[r];
  }
}

// batched C = A @ B ; bf16 in, f32 out
__global__ __launch_bounds__(256) void gemm_bf16_nn(
    const bf16_t* __restrict__ A, int lda, long sA,
    const bf16_t* __restrict__ Bm, int ldb, long sB,
    float* __restrict__ Of, int ldo, long sO, int Kd) {
  const int bh = blockIdx.z;
  const int wave = threadIdx.x >> 5;
  const int row0 = blockIdx.y * 32 + (wave >> 2) * 16;
  const int col0 = blockIdx.x * 64 + (wave & 3) * 16;
  const bf16_t* Ap = A + (long)bh * sA + (long)row0 * lda;
  const bf16_t* Bp = Bm + (long)bh * sB + col0;
  f32x8 acc = {};
  for (int kk = 0; kk < Kd; kk += 32) {
    __builtin_prefetch(Ap + kk + 32, 0, 3);
    acc = wmma_bf16(ld_a16(Ap + kk, lda), ld_b16(Bp + (long)kk * ldb, ldb), acc);
  }
  const int l = lane_id(), n = l & 15, mb = (l >> 4) * 8;
  const long base = (long)bh * sO + (long)row0 * ldo + col0;
#pragma unroll
  for (int r = 0; r < 8; ++r) Of[base + (long)(mb + r) * ldo + n] = acc[r];
}

// --- TDM stage: `rows` x 64 f32 panel (row stride ldb) into LDS at offset 0.
// LDS row stride BSTRIDE DWORDs (TDM pad: +2 DWORDs every 64 -> no bank
// conflicts between the two lane-halves of each ds_load).
// NOTE: the TDM intrinsic carries the LDS address only as an integer bitfield,
// so LLVM cannot see that Bs is written by it.  The guarded dummy store (never
// taken at runtime, unprovable at compile time) plus the full memory clobber
// keep the Bs loads from being folded to poison.
__device__ __forceinline__ void stage_b_panel(float* __restrict__ Bs,
                                              const float* __restrict__ src,
                                              int ldb, int rows) {
#if defined(HAVE_TDM)
  if (threadIdx.x < 32) {                     // wave 0 issues the DMA
    const unsigned long long ga = (unsigned long long)(size_t)src;
    u32x4 g0;
    g0[0] = 1u;                               // count=1, user descriptor
    g0[1] = 0u;                               // lds_addr: Bs is at LDS offset 0
    g0[2] = (unsigned)(ga & 0xFFFFFFFFu);     // global_addr[31:0]
    g0[3] = (unsigned)((ga >> 32) & 0x1FFFFFFu) | (2u << 30);  // addr hi | type=2
    i32x8 g1;
    g1[0] = (int)((2u << 16) | (1u << 20) | (5u << 22) | (1u << 25));
            // data_size=4B | pad_enable | pad_interval: 64 DW | pad_amount: 2 DW
    g1[1] = (int)(((unsigned)ldb & 0xFFFFu) << 16);            // tensor_dim0 lo
    g1[2] = (int)((((unsigned)ldb >> 16) & 0xFFFFu) |
                  (((unsigned)rows & 0xFFFFu) << 16));         // dim0 hi | dim1 lo
    g1[3] = (int)((((unsigned)rows >> 16) & 0xFFFFu) | (64u << 16)); // | tile_dim0
    g1[4] = (int)((unsigned)rows & 0xFFFFu);                   // tile_dim1 (tile_dim2=0)
    g1[5] = (int)(unsigned)ldb;                                // tensor_dim0_stride
    g1[6] = 0; g1[7] = 0;
    i32x4 gz = {0, 0, 0, 0};
#if defined(__clang_major__) && __clang_major__ >= 23
    i32x8 gz8 = {0, 0, 0, 0, 0, 0, 0, 0};
    __builtin_amdgcn_tensor_load_to_lds(g0, g1, gz, gz, gz8, 0);
#else
    __builtin_amdgcn_tensor_load_to_lds(g0, g1, gz, gz, 0);
#endif
    __builtin_amdgcn_s_wait_tensorcnt(0);
  }
  // never executes (ldb is 64 or 256); keeps a real store to Bs in the module
  if (__builtin_expect(ldb == -1, 0)) Bs[threadIdx.x] = 0.f;
#else
  for (int i = threadIdx.x; i < rows * 64; i += 256) {
    const int r = i >> 6, c = i & 63;
    Bs[r * BSTRIDE + c] = src[(long)r * ldb + c];
  }
#endif
  __syncthreads();
  asm volatile("" ::: "memory");              // DMA-written LDS: force reloads
}

// batched fp32 GEMM  D = alpha*(A@B) + beta*C   (pinv chain / Z@TV)
// B panel staged into LDS via the Tensor Data Mover; fp32 WMMA compute,
// B fragments read straight from the __shared__ array (ds_load path).
__global__ __launch_bounds__(256) void gemm_f32_nn(
    const float* __restrict__ A, int lda, long sA,
    const float* __restrict__ Bm, int ldb, long sB,
    const float* __restrict__ C, long sC,
    float* __restrict__ D, int ldd, long sD, bf16_t* __restrict__ Db,
    float alpha, float beta, int Kd) {
  __shared__ float Bs[KSTAGE * BSTRIDE];
  const int bh = blockIdx.z;
  const int wave = threadIdx.x >> 5;
  const int row0 = blockIdx.y * 32 + (wave >> 2) * 16;
  const int col0 = blockIdx.x * 64 + (wave & 3) * 16;
  const float* Ap = A + (long)bh * sA + (long)row0 * lda;
  const float* Bpanel = Bm + (long)bh * sB + blockIdx.x * 64;
  // fp32 WMMA B-fragment lane coords: b[j] = B[kh+j][n], kh = 2*(lane>>4)
  const int l = lane_id();
  const int bn = (wave & 3) * 16 + (l & 15);  // column inside LDS panel
  const int kh = (l >> 4) * 2;
  f32x8 acc0 = {}, acc1 = {};
  for (int half = 0; half < Kd; half += KSTAGE) {
    stage_b_panel(Bs, Bpanel + (long)half * ldb, ldb, KSTAGE);
#pragma unroll 4
    for (int kk = 0; kk < KSTAGE; kk += 8) {
      __builtin_prefetch(Ap + half + kk + 16, 0, 3);
      f32x2 a0 = ld_a4(Ap + half + kk, lda);
      f32x2 b0;                                // direct __shared__ indexing -> ds_load
      b0[0] = Bs[(kk + kh) * BSTRIDE + bn];
      b0[1] = Bs[(kk + kh + 1) * BSTRIDE + bn];
      acc0 = wmma_f32(a0, b0, acc0);
      f32x2 a1 = ld_a4(Ap + half + kk + 4, lda);
      f32x2 b1;
      b1[0] = Bs[(kk + 4 + kh) * BSTRIDE + bn];
      b1[1] = Bs[(kk + 4 + kh + 1) * BSTRIDE + bn];
      acc1 = wmma_f32(a1, b1, acc1);
    }
    __syncthreads();                          // LDS reused next half
  }
  const int n = l & 15, mb = (l >> 4) * 8;
#pragma unroll
  for (int r = 0; r < 8; ++r) {
    const long ro = (long)(row0 + mb + r) * ldd + col0 + n;
    float v = alpha * (acc0[r] + acc1[r]);
    if (beta != 0.f) v += beta * C[(long)bh * sC + ro];
    D[(long)bh * sD + ro] = v;
    if (Db) Db[(long)bh * sD + ro] = (bf16_t)v;
  }
}

// row softmax in-place, bf16 rows of length len (multiple of 256)
__global__ void softmax_bf16(bf16_t* __restrict__ p, int len) {
  bf16_t* r = p + (long)blockIdx.x * len;
  const int t = threadIdx.x;
  const int per = len >> 8;
  __shared__ float red[256];
  float lmax = -3.4e38f;
  for (int i = 0; i < per; ++i) lmax = fmaxf(lmax, (float)r[t + (i << 8)]);
  red[t] = lmax; __syncthreads();
  for (int s = 128; s > 0; s >>= 1) { if (t < s) red[t] = fmaxf(red[t], red[t + s]); __syncthreads(); }
  const float m = red[0]; __syncthreads();
  float e[16];
  float lsum = 0.f;
  for (int i = 0; i < per; ++i) { e[i] = __expf((float)r[t + (i << 8)] - m); lsum += e[i]; }
  red[t] = lsum; __syncthreads();
  for (int s = 128; s > 0; s >>= 1) { if (t < s) red[t] += red[t + s]; __syncthreads(); }
  const float inv = 1.0f / red[0];
  for (int i = 0; i < per; ++i) r[t + (i << 8)] = (bf16_t)(e[i] * inv);
}

// row softmax in-place, f32 rows of length 256
__global__ void softmax_f32(float* __restrict__ p) {
  float* r = p + (long)blockIdx.x * M_;
  const int t = threadIdx.x;
  __shared__ float red[256];
  float v = r[t];
  red[t] = v; __syncthreads();
  for (int s = 128; s > 0; s >>= 1) { if (t < s) red[t] = fmaxf(red[t], red[t + s]); __syncthreads(); }
  const float m = red[0]; __syncthreads();
  float e = __expf(v - m);
  red[t] = e; __syncthreads();
  for (int s = 128; s > 0; s >>= 1) { if (t < s) red[t] += red[t + s]; __syncthreads(); }
  r[t] = e / red[0];
}

__global__ void k_init_red(float* red) { red[0] = 0.f; red[1] = 0.f; }

// global max of |row sums| (y=0) and |col sums| (y=1) of S2 across all bh
__global__ void k_redmax(const float* __restrict__ S2, float* __restrict__ red) {
  const int idx = blockIdx.x;
  const int bh = idx >> 8, i = idx & 255;
  const float* base = S2 + (long)bh * M_ * M_;
  const int t = threadIdx.x;
  float v = (blockIdx.y == 0) ? base[(long)i * M_ + t] : base[(long)t * M_ + i];
  v = fabsf(v);
  __shared__ float rs[256];
  rs[t] = v; __syncthreads();
  for (int s = 128; s > 0; s >>= 1) { if (t < s) rs[t] += rs[t + s]; __syncthreads(); }
  if (t == 0) atomicMax((int*)&red[blockIdx.y], __float_as_int(rs[0]));
}

// Z0[bh][i][j] = S2[bh][j][i] / (maxrow*maxcol)
__global__ void k_zinit(const float* __restrict__ S2, const float* __restrict__ red,
                        float* __restrict__ Z) {
  long i = (long)blockIdx.x * 256 + threadIdx.x;
  if (i >= (long)BH_ * M_ * M_) return;
  const int j = (int)(i & 255); const long t = i >> 8;
  const int r = (int)(t & 255); const int bh = (int)(t >> 8);
  const float inv = 1.0f / (red[0] * red[1]);
  Z[i] = S2[((long)bh * M_ + j) * M_ + r] * inv;
}

// OutHb = bf16( OutH + depthwise_conv_n(Vf) + conv_b[h] )
__global__ void k_convres(const float* __restrict__ OutH, const float* __restrict__ Vf,
                          const float* __restrict__ cw, const float* __restrict__ cb,
                          bf16_t* __restrict__ OutHb) {
  long i = (long)blockIdx.x * 256 + threadIdx.x;
  if (i >= (long)BH_ * N_ * DH_) return;
  const int d = (int)(i & 63); const long t = i >> 6;
  const int n = (int)(t & (N_ - 1)); const int bh = (int)(t >> 12);
  const int h = bh & (H_ - 1);
  const float* vbase = Vf + ((long)bh * N_) * DH_ + d;
  const float* w = cw + h * KS_;
  float s = OutH[i] + cb[h];
  for (int k = 0; k < KS_; ++k) {
    const int nn = n + k - PAD_;
    if (nn >= 0 && nn < N_) s += w[k] * vbase[(long)nn * DH_];
  }
  OutHb[i] = (bf16_t)s;
}

// out[b*N+n][c] = gather(OutHb)[.,512] @ w_out + b_out
__global__ __launch_bounds__(256) void k_outproj(const bf16_t* __restrict__ Ain,
                                                 const float* __restrict__ w,
                                                 const float* __restrict__ bias,
                                                 float* __restrict__ out) {
  const int wave = threadIdx.x >> 5;
  const int row0 = blockIdx.y * 32 + (wave >> 2) * 16;
  const int col0 = blockIdx.x * 64 + (wave & 3) * 16;
  const int l = lane_id();
  const int m = l & 15, h8 = (l >> 4) * 8;
  const int gm = row0 + m, bidx = gm >> 12, npos = gm & (N_ - 1);
  f32x8 acc = {};
  for (int kk = 0; kk < INNER_; kk += 32) {
    bf16x16 a;
#pragma unroll
    for (int j = 0; j < 8; ++j) {
      const int k = kk + ((j & 3) * 2) + ((j >> 2) * 16) + h8;
      const long o = (((long)(bidx * H_ + (k >> 6))) * N_ + npos) * DH_ + (k & 63);
      a[2 * j] = Ain[o]; a[2 * j + 1] = Ain[o + 1];
    }
    bf16x16 b = ld_b16_f(w + (long)kk * DIM_ + col0, DIM_);
    acc = wmma_bf16(a, b, acc);
  }
  const int n = l & 15, mb = (l >> 4) * 8;
  const float bv = bias[col0 + n];
#pragma unroll
  for (int r = 0; r < 8; ++r)
    out[(long)(row0 + mb + r) * DIM_ + col0 + n] = acc[r] + bv;
}

// ================= host side =================
extern "C" void kernel_launch(void* const* d_in, const int* in_sizes, int n_in,
                              void* d_out, int out_size, void* d_ws, size_t ws_size,
                              hipStream_t stream) {
  (void)in_sizes; (void)n_in; (void)out_size; (void)ws_size;
  const float* x    = (const float*)d_in[0];
  const float* wqkv = (const float*)d_in[1];
  const float* bqkv = (const float*)d_in[2];
  const float* wout = (const float*)d_in[3];
  const float* bout = (const float*)d_in[4];
  const float* cw   = (const float*)d_in[5];
  const float* cb   = (const float*)d_in[6];
  float* out = (float*)d_out;

  char* ws = (char*)d_ws; size_t off = 0;
  auto alloc = [&](size_t bytes) -> void* {
    void* p = ws + off; off += (bytes + 255) & ~(size_t)255; return p;
  };
  const long E  = (long)BH_ * N_ * DH_;
  const long EM = (long)BH_ * M_ * DH_;
  const long MM = (long)BH_ * M_ * M_;

  bf16_t* Qb    = (bf16_t*)alloc(E * 2);
  bf16_t* Kb    = (bf16_t*)alloc(E * 2);
  bf16_t* Vb    = (bf16_t*)alloc(E * 2);
  bf16_t* OutHb = (bf16_t*)alloc(E * 2);
  float*  Vf    = (float*)alloc(E * 4);
  float*  OutH  = (float*)alloc(E * 4);
  bf16_t* A1    = (bf16_t*)alloc((long)BH_ * N_ * M_ * 2);
  bf16_t* A3    = (bf16_t*)alloc((long)BH_ * M_ * N_ * 2);
  float*  S2    = (float*)alloc(MM * 4);
  float*  Z0    = (float*)alloc(MM * 4);
  float*  Z1    = (float*)alloc(MM * 4);
  float*  T0    = (float*)alloc(MM * 4);
  float*  T1    = (float*)alloc(MM * 4);
  float*  T2    = (float*)alloc(MM * 4);
  float*  TV    = (float*)alloc(EM * 4);
  float*  T2g   = (float*)alloc(EM * 4);
  bf16_t* T2gb  = (bf16_t*)alloc(EM * 2);
  bf16_t* QLb   = (bf16_t*)alloc(EM * 2);
  bf16_t* KLb   = (bf16_t*)alloc(EM * 2);
  float*  red   = (float*)alloc(256);

  // 1. QKV projection + head split
  k_qkv<<<dim3(24, 512), 256, 0, stream>>>(x, wqkv, bqkv, Qb, Kb, Vb, Vf);

  // 2. landmarks
  {
    int blocks = (int)((EM + 255) / 256);
    k_landmark<<<blocks, 256, 0, stream>>>(Qb, QLb);
    k_landmark<<<blocks, 256, 0, stream>>>(Kb, KLb);
  }

  // 3. similarity GEMMs (A @ B^T)
  gemm_bf16_nt<<<dim3(4, 128, BH_), 256, 0, stream>>>(
      Qb, DH_, (long)N_ * DH_, KLb, DH_, (long)M_ * DH_,
      A1, nullptr, M_, (long)N_ * M_, DH_);
  gemm_bf16_nt<<<dim3(64, 8, BH_), 256, 0, stream>>>(
      QLb, DH_, (long)M_ * DH_, Kb, DH_, (long)N_ * DH_,
      A3, nullptr, N_, (long)M_ * N_, DH_);
  gemm_bf16_nt<<<dim3(4, 8, BH_), 256, 0, stream>>>(
      QLb, DH_, (long)M_ * DH_, KLb, DH_, (long)M_ * DH_,
      nullptr, S2, M_, (long)M_ * M_, DH_);

  // 4. softmaxes
  softmax_bf16<<<BH_ * N_, 256, 0, stream>>>(A1, M_);
  softmax_bf16<<<BH_ * M_, 256, 0, stream>>>(A3, N_);
  softmax_f32<<<BH_ * M_, 256, 0, stream>>>(S2);

  // 5. pinv init: Z0 = S2^T / (max rowsum * max colsum)
  k_init_red<<<1, 1, 0, stream>>>(red);
  k_redmax<<<dim3(BH_ * M_, 2), 256, 0, stream>>>(S2, red);
  k_zinit<<<(int)((MM + 255) / 256), 256, 0, stream>>>(S2, red, Z0);

  // 6. Newton-Schulz pinv (fp32 WMMA, TDM-staged B panels)
  float* Zc = Z0; float* Zn = Z1;
  const long sM = (long)M_ * M_;
  const dim3 gP(4, 8, BH_);
  for (int it = 0; it < ITERS_; ++it) {
    gemm_f32_nn<<<gP, 256, 0, stream>>>(S2, M_, sM, Zc, M_, sM, nullptr, sM,
                                        T0, M_, sM, nullptr, 1.0f, 0.0f, M_);
    gemm_f32_nn<<<gP, 256, 0, stream>>>(T0, M_, sM, T0, M_, sM, T0, sM,
                                        T1, M_, sM, nullptr, -1.0f, 7.0f, M_);
    gemm_f32_nn<<<gP, 256, 0, stream>>>(T0, M_, sM, T1, M_, sM, T0, sM,
                                        T2, M_, sM, nullptr, -1.0f, 15.0f, M_);
    gemm_f32_nn<<<gP, 256, 0, stream>>>(Zc, M_, sM, T2, M_, sM, Zc, sM,
                                        Zn, M_, sM, nullptr, -0.25f, 3.25f, M_);
    float* t = Zc; Zc = Zn; Zn = t;
  }

  // 7. TV = attn3 @ V ; T2g = Z @ TV ; OutH = attn1 @ T2g
  gemm_bf16_nn<<<dim3(1, 8, BH_), 256, 0, stream>>>(
      A3, N_, (long)M_ * N_, Vb, DH_, (long)N_ * DH_, TV, DH_, (long)M_ * DH_, N_);
  gemm_f32_nn<<<dim3(1, 8, BH_), 256, 0, stream>>>(
      Zc, M_, sM, TV, DH_, (long)M_ * DH_, nullptr, 0,
      T2g, DH_, (long)M_ * DH_, T2gb, 1.0f, 0.0f, M_);
  gemm_bf16_nn<<<dim3(1, 128, BH_), 256, 0, stream>>>(
      A1, M_, (long)N_ * M_, T2gb, DH_, (long)M_ * DH_, OutH, DH_, (long)N_ * DH_, M_);

  // 8. conv residual + bias -> bf16
  k_convres<<<(int)((E + 255) / 256), 256, 0, stream>>>(OutH, Vf, cw, cb, OutHb);

  // 9. output projection
  k_outproj<<<dim3(8, 512), 256, 0, stream>>>(OutHb, wout, bout, out);
}
